// GRIPPlusPlus_23347442221696
// MI455X (gfx1250) — compile-verified, matrix operand order checked by reading, not run on hardware
//
#include <hip/hip_runtime.h>
#include <math.h>

// ---------------- model constants ----------------
#define BATCH 16
#define THIST 20
#define TFUT  30
#define HID   128
#define NNBR  127
#define NNODE 128   // 1 ego + 127 nbr
#define NMODE 6
#define NSEQ  2048  // 16 ego + 2032 nbr, grouped as 128 groups of 16
#define NGRP  128

typedef __attribute__((ext_vector_type(16))) _Float16 v16h;
typedef __attribute__((ext_vector_type(8)))  float    v8f;

union VF8 { v8f v; float f[8]; };

__device__ __forceinline__ v8f wmma_f16(v16h a, v16h b, v8f c) {
  // D(f32 16x16) = A(f16 16x32) * B(f16 32x16) + C
  return __builtin_amdgcn_wmma_f32_16x16x32_f16(false, a, false, b, (short)0, c, false, false);
}

// A-fragment (16x32 f16) element index for (row m, k): lane*16 + half
__device__ __forceinline__ int afrag_idx(int m, int k) {
  int lane = (m & 15) | (((k >> 3) & 1) << 4);
  int h    = (k & 7) | ((k & 16) >> 1);
  return (lane << 4) | h;
}
// B-fragment (32x16 f16) element index for (k, col n)
__device__ __forceinline__ int bfrag_idx(int k, int n) {
  int lane = (n & 15) | (((k >> 3) & 1) << 4);
  int h    = (k & 7) | ((k & 16) >> 1);
  return (lane << 4) | h;
}
// inverse of afrag_idx: element e -> (row, k)
__device__ __forceinline__ void inv_frag(int e, int& row, int& k) {
  int lane = e >> 4, h = e & 15;
  row = lane & 15;
  k = (h & 7) | (((lane >> 4) & 1) << 3) | ((h & 8) << 1);
}
__device__ __forceinline__ v8f vz() { VF8 r; for (int i = 0; i < 8; ++i) r.f[i] = 0.f; return r.v; }
__device__ __forceinline__ float sigmoidf_(float x) { return 1.f / (1.f + __expf(-x)); }

// ---------------- weight packing ----------------
// Pack W (stored row-major [N][ld], used as x@W^T => B[k][n]=W[n][koff+k]) into B-frags.
__global__ __launch_bounds__(256) void pack_b(const float* W, int ld, int koff, int K, int N,
                                              _Float16* dst, int NtTot, int nt0, int NtUsed) {
  int f = blockIdx.x;
  int kt = f / NtUsed, nt = f % NtUsed;
  _Float16* d = dst + ((size_t)(kt * NtTot) + nt0 + nt) * 512;
  for (int i = threadIdx.x; i < 512; i += 256) {
    int lane = i >> 4, h = i & 15;
    int n = lane & 15;
    int k = (h & 7) | (((lane >> 4) & 1) << 3) | ((h & 8) << 1);
    int gk = kt * 32 + k, gn = nt * 16 + n;
    float v = (gk < K && gn < N) ? W[(size_t)gn * ld + koff + gk] : 0.f;
    d[i] = (_Float16)v;
  }
}

// conv2 weights (128,64,3): GEMM K index = ktap*64 + ic  => custom gather
__global__ __launch_bounds__(256) void pack_conv2(const float* W, _Float16* dst) {
  int f = blockIdx.x;            // 6 k-tiles * 8 n-tiles
  int kt = f / 8, nt = f % 8;
  _Float16* d = dst + ((size_t)(kt * 8 + nt)) * 512;
  for (int i = threadIdx.x; i < 512; i += 256) {
    int lane = i >> 4, h = i & 15;
    int n = lane & 15;
    int k = (h & 7) | (((lane >> 4) & 1) << 3) | ((h & 8) << 1);
    int gk = kt * 32 + k, gn = nt * 16 + n;
    float v = 0.f;
    if (gk < 192) { int ktap = gk >> 6, ic = gk & 63; v = W[((size_t)gn * 64 + ic) * 3 + ktap]; }
    d[i] = (_Float16)v;
  }
}

// ---------------- encoder: conv1 + conv2 (WMMA) ----------------
struct EncConvP {
  const float* c1W; const float* c1b;
  const float* g1;  const float* b1; const float* rm1; const float* rv1;
  const float* c2b;
  const float* g2;  const float* b2; const float* rm2; const float* rv2;
  const _Float16* pkc2;
};

__global__ __launch_bounds__(256) void conv_encoder(const float* history, const float* neighbors,
                                                    EncConvP ego, EncConvP nbr, _Float16* xfrags) {
  int g = blockIdx.x;                   // group of 16 sequences; g==0 -> ego
  int tid = threadIdx.x, lane = tid & 31, wv = tid >> 5;
  EncConvP P = (g == 0) ? ego : nbr;
  __shared__ float xin[16][THIST + 2][2];
  __shared__ alignas(64) _Float16 y1h[16][THIST + 2][64];
  __shared__ alignas(64) _Float16 afr[6 * 512];

  // stage raw input with zero time-padding
  for (int i = tid; i < 16 * 22 * 2; i += 256) {
    int m = i / 44, r = i % 44, t = r >> 1, c = r & 1;
    float v = 0.f; int tt = t - 1;
    if (tt >= 0 && tt < THIST) {
      if (g == 0) v = history[((size_t)m * THIST + tt) * 2 + c];
      else { size_t s = (size_t)(g - 1) * 16 + m; v = neighbors[(s * THIST + tt) * 2 + c]; }
    }
    xin[m][t][c] = v;
  }
  __syncthreads();
  // conv1 + bn1 + relu (store f16, with t padding rows)
  for (int i = tid; i < 16 * 64; i += 256) { int m = i >> 6, c = i & 63; y1h[m][0][c] = (_Float16)0.f; y1h[m][21][c] = (_Float16)0.f; }
  for (int i = tid; i < 16 * THIST * 64; i += 256) {
    int m = i / (THIST * 64), r = i % (THIST * 64), t = r >> 6, c = r & 63;
    float acc = P.c1b[c];
    for (int kt = 0; kt < 3; ++kt)
      acc += xin[m][t + kt][0] * P.c1W[(c * 2 + 0) * 3 + kt] + xin[m][t + kt][1] * P.c1W[(c * 2 + 1) * 3 + kt];
    float inv = P.g1[c] * rsqrtf(P.rv1[c] + 1e-5f);
    float y = (acc - P.rm1[c]) * inv + P.b1[c];
    y1h[m][t + 1][c] = (_Float16)fmaxf(y, 0.f);
  }
  __syncthreads();
  // conv2 as GEMM per t: A = im2col(16 x 192), B = packed weights (192 x 128)
  for (int t = 0; t < THIST; ++t) {
    for (int i = tid; i < 6 * 512; i += 256) {
      int kt = i >> 9, e = i & 511, m, k; inv_frag(e, m, k);
      int K = kt * 32 + k, ktap = K >> 6, ic = K & 63;
      afr[i] = y1h[m][t + ktap][ic];
    }
    __syncthreads();
    VF8 acc; acc.v = vz();
    for (int kk = 0; kk < 6; ++kk) {
      v16h a = ((const v16h*)(afr + kk * 512))[lane];
      const _Float16* bp = P.pkc2 + (size_t)(kk * 8 + wv) * 512;
      __builtin_prefetch(bp + 512, 0, 1);
      v16h b = ((const v16h*)bp)[lane];
      acc.v = wmma_f16(a, b, acc.v);
    }
    for (int i2 = 0; i2 < 8; ++i2) {
      int m = i2 + ((lane >> 4) << 3), n = lane & 15;
      int c = wv * 16 + n;
      float inv = P.g2[c] * rsqrtf(P.rv2[c] + 1e-5f);
      float y = (acc.f[i2] + P.c2b[c] - P.rm2[c]) * inv + P.b2[c];
      y = fmaxf(y, 0.f);
      int kt = c >> 5, dk = c & 31;
      size_t fi = ((size_t)(g * THIST + t) * 4 + kt) * 512 + afrag_idx(m, dk);
      xfrags[fi] = (_Float16)y;        // already laid out as GRU A-fragment
    }
    __syncthreads();
  }
}

// ---------------- GRU (WMMA) ----------------
struct GruP { const _Float16* pkWih; const _Float16* pkWhh; const float* bih; const float* bhh; };

__global__ __launch_bounds__(256) void gru_kernel(GruP ego, GruP nbr, const _Float16* xfrags, float* hT) {
  int g = blockIdx.x;
  int tid = threadIdx.x, lane = tid & 31, wv = tid >> 5;
  GruP P = (g == 0) ? ego : nbr;
  __shared__ float sgi[16][384];
  __shared__ float sgh[16][384];
  __shared__ float sh[16][128];
  __shared__ alignas(64) _Float16 hfr[4 * 512];
  for (int i = tid; i < 16 * 128; i += 256) { int m = i >> 7, d = i & 127; sh[m][d] = 0.f; }
  for (int i = tid; i < 4 * 512; i += 256) hfr[i] = (_Float16)0.f;
  __syncthreads();

  for (int t = 0; t < THIST; ++t) {
    for (int j = wv; j < 24; j += 8) {
      VF8 ai, ah; ai.v = vz(); ah.v = vz();
      for (int kk = 0; kk < 4; ++kk) {
        const _Float16* xp = xfrags + ((size_t)(g * THIST + t) * 4 + kk) * 512;
        v16h a = ((const v16h*)xp)[lane];
        const _Float16* bp = P.pkWih + (size_t)(kk * 24 + j) * 512;
        __builtin_prefetch(bp + 512, 0, 1);
        v16h b = ((const v16h*)bp)[lane];
        ai.v = wmma_f16(a, b, ai.v);
        v16h ha = ((const v16h*)(hfr + kk * 512))[lane];
        v16h b2 = ((const v16h*)(P.pkWhh + (size_t)(kk * 24 + j) * 512))[lane];
        ah.v = wmma_f16(ha, b2, ah.v);
      }
      for (int i2 = 0; i2 < 8; ++i2) {
        int m = i2 + ((lane >> 4) << 3), n = lane & 15;
        sgi[m][j * 16 + n] = ai.f[i2];
        sgh[m][j * 16 + n] = ah.f[i2];
      }
    }
    __syncthreads();
    for (int i = tid; i < 16 * 128; i += 256) {
      int m = i >> 7, d = i & 127;
      float ir = sgi[m][d]       + P.bih[d];
      float iz = sgi[m][128 + d] + P.bih[128 + d];
      float in_= sgi[m][256 + d] + P.bih[256 + d];
      float hr = sgh[m][d]       + P.bhh[d];
      float hz = sgh[m][128 + d] + P.bhh[128 + d];
      float hn = sgh[m][256 + d] + P.bhh[256 + d];
      float r = sigmoidf_(ir + hr);
      float z = sigmoidf_(iz + hz);
      float n2 = tanhf(in_ + r * hn);
      float hnew = (1.f - z) * n2 + z * sh[m][d];
      sh[m][d] = hnew;
      hfr[(d >> 5) * 512 + afrag_idx(m, d & 31)] = (_Float16)hnew;
    }
    __syncthreads();
  }
  for (int i = tid; i < 16 * 128; i += 256) {
    int m = i >> 7, d = i & 127;
    hT[((size_t)g * 16 + m) * 128 + d] = sh[m][d];
  }
}

// ---------------- assemble node features ----------------
__global__ void assemble_nodes(const float* hT, const int* mask, const float* te, float* nf) {
  int i = blockIdx.x * blockDim.x + threadIdx.x;
  if (i >= BATCH * NNODE * HID) return;
  int d = i & 127, node = (i >> 7) & 127, b = i >> 14;
  float v;
  if (node == 0) v = hT[(size_t)b * 128 + d] + te[d];
  else {
    float mf = (float)(mask[b * NNBR + node - 1]);
    v = hT[((size_t)16 + (size_t)b * NNBR + node - 1) * 128 + d] * mf + te[128 + d];
  }
  nf[i] = v;
}

// ---------------- GCN: fused projections (node | Pi | Pj), one N=384 GEMM ----------------
__global__ __launch_bounds__(256) void gcn_proj(const float* x, const _Float16* pkproj,
                                                const float* node_b, const float* e1_b,
                                                _Float16* HnFr, float* Pi, float* Pj) {
  int blk = blockIdx.x;             // 128 = 16 batches * 8 node-tiles
  int b = blk >> 3, it = blk & 7;
  int tid = threadIdx.x, lane = tid & 31, wv = tid >> 5;
  __shared__ alignas(64) _Float16 afr[4 * 512];
  for (int i = tid; i < 4 * 512; i += 256) {
    int kt = i >> 9, e = i & 511, m, k; inv_frag(e, m, k);
    int row = it * 16 + m;
    afr[i] = (_Float16)x[((size_t)b * NNODE + row) * 128 + kt * 32 + k];
  }
  __syncthreads();
  for (int j = wv; j < 24; j += 8) {
    VF8 acc; acc.v = vz();
    for (int kk = 0; kk < 4; ++kk) {
      v16h a = ((const v16h*)(afr + kk * 512))[lane];
      const _Float16* bp = pkproj + (size_t)(kk * 24 + j) * 512;
      __builtin_prefetch(bp + 512, 0, 1);
      v16h bb = ((const v16h*)bp)[lane];
      acc.v = wmma_f16(a, bb, acc.v);
    }
    for (int i2 = 0; i2 < 8; ++i2) {
      int m = i2 + ((lane >> 4) << 3), n = lane & 15;
      int row = it * 16 + m;
      if (j < 8) {                   // node-projection -> Hn as B-frags for msg GEMM
        int d = j * 16 + n;
        float v = acc.f[i2] + node_b[d];
        HnFr[(((size_t)b * 4 + (row >> 5)) * 8 + j) * 512 + bfrag_idx(row & 31, n)] = (_Float16)v;
      } else if (j < 16) {           // Pi (carries e1 bias)
        int d = (j - 8) * 16 + n;
        Pi[((size_t)b * NNODE + row) * 128 + d] = acc.f[i2] + e1_b[d];
      } else {                       // Pj
        int d = (j - 16) * 16 + n;
        Pj[((size_t)b * NNODE + row) * 128 + d] = acc.f[i2];
      }
    }
  }
}

// ---------------- GCN: edge scores + masked softmax (linearized e1) ----------------
__global__ __launch_bounds__(128) void gcn_edge(const float* Pi, const float* Pj,
                                                const float* e1W, const float* e2W, const float* e2b,
                                                const int* mask, const float* history, const float* neighbors,
                                                float* wout) {
  int b = blockIdx.x >> 7, i = blockIdx.x & 127;
  int j = threadIdx.x;
  __shared__ float spi[128];
  __shared__ float red[128];
  spi[j] = Pi[((size_t)b * NNODE + i) * 128 + j];
  __syncthreads();
  float pix, piy, pjx, pjy;
  if (i == 0) { pix = history[(b * THIST + 19) * 2 + 0]; piy = history[(b * THIST + 19) * 2 + 1]; }
  else { size_t o = (((size_t)b * NNBR + (i - 1)) * THIST + 19) * 2; pix = neighbors[o]; piy = neighbors[o + 1]; }
  if (j == 0) { pjx = history[(b * THIST + 19) * 2 + 0]; pjy = history[(b * THIST + 19) * 2 + 1]; }
  else { size_t o = (((size_t)b * NNBR + (j - 1)) * THIST + 19) * 2; pjx = neighbors[o]; pjy = neighbors[o + 1]; }
  float rx = pix - pjx, ry = piy - pjy;
  float dist = sqrtf(rx * rx + ry * ry + 1e-12f);
  int mi = (i == 0) ? 1 : (mask[b * NNBR + i - 1] != 0);
  int mj = (j == 0) ? 1 : (mask[b * NNBR + j - 1] != 0);
  int keep = mi && mj;
  float s = -1e30f;
  if (keep) {
    float acc = e2b[0];
    const float* pjr = Pj + ((size_t)b * NNODE + j) * 128;
    for (int d = 0; d < 128; ++d) {
      const float* wr = e1W + (size_t)d * 259;
      float ev = spi[d] + pjr[d] + rx * wr[256] + ry * wr[257] + dist * wr[258];
      acc += fmaxf(ev, 0.f) * e2W[d];
    }
    s = acc;
  }
  red[j] = s; __syncthreads();
  for (int o = 64; o > 0; o >>= 1) { if (j < o) red[j] = fmaxf(red[j], red[j + o]); __syncthreads(); }
  float mx = red[0]; __syncthreads();
  float e = keep ? __expf(s - mx) : 0.f;
  red[j] = e; __syncthreads();
  for (int o = 64; o > 0; o >>= 1) { if (j < o) red[j] += red[j + o]; __syncthreads(); }
  float den = red[0]; if (den == 0.f) den = 1.f;
  wout[((size_t)b * NNODE + i) * 128 + j] = e / den;
}

// ---------------- GCN: message GEMM + agg GEMM + residual + LN (fused, WMMA) ----------------
__global__ __launch_bounds__(256) void gcn_msg(const float* wsm, const _Float16* HnFr,
                                               const _Float16* pkagg, const float* agg_b,
                                               const float* x, const float* lng, const float* lnb,
                                               float* out) {
  int blk = blockIdx.x;             // 16 batches * 8 row-tiles
  int b = blk >> 3, it = blk & 7;
  int tid = threadIdx.x, lane = tid & 31, wv = tid >> 5;
  __shared__ alignas(64) _Float16 afr[4 * 512];
  __shared__ alignas(64) _Float16 mfr[4 * 512];
  __shared__ float sy[16][128];
  __shared__ float smu[16], srs[16];
  for (int i = tid; i < 4 * 512; i += 256) {
    int kt = i >> 9, e = i & 511, m, k; inv_frag(e, m, k);
    int row = it * 16 + m;
    afr[i] = (_Float16)wsm[((size_t)b * NNODE + row) * 128 + kt * 32 + k];
  }
  __syncthreads();
  VF8 acc; acc.v = vz();
  for (int kk = 0; kk < 4; ++kk) {
    v16h a = ((const v16h*)(afr + kk * 512))[lane];
    v16h bb = ((const v16h*)(HnFr + (((size_t)b * 4 + kk) * 8 + wv) * 512))[lane];
    acc.v = wmma_f16(a, bb, acc.v);
  }
  for (int i2 = 0; i2 < 8; ++i2) {
    int m = i2 + ((lane >> 4) << 3), n = lane & 15;
    int c = wv * 16 + n;
    mfr[(c >> 5) * 512 + afrag_idx(m, c & 31)] = (_Float16)acc.f[i2];
  }
  __syncthreads();
  VF8 a2; a2.v = vz();
  for (int kk = 0; kk < 4; ++kk) {
    v16h a = ((const v16h*)(mfr + kk * 512))[lane];
    const _Float16* bp = pkagg + (size_t)(kk * 8 + wv) * 512;
    __builtin_prefetch(bp + 512, 0, 1);
    v16h bb = ((const v16h*)bp)[lane];
    a2.v = wmma_f16(a, bb, a2.v);
  }
  for (int i2 = 0; i2 < 8; ++i2) {
    int m = i2 + ((lane >> 4) << 3), n = lane & 15;
    int c = wv * 16 + n, row = it * 16 + m;
    float v = fmaxf(a2.f[i2] + agg_b[c], 0.f) + x[((size_t)b * NNODE + row) * 128 + c];
    sy[m][c] = v;
  }
  __syncthreads();
  if (tid < 16) {
    float mu = 0.f;
    for (int d = 0; d < 128; ++d) mu += sy[tid][d];
    mu *= (1.f / 128.f);
    float var = 0.f;
    for (int d = 0; d < 128; ++d) { float t = sy[tid][d] - mu; var += t * t; }
    var *= (1.f / 128.f);
    smu[tid] = mu; srs[tid] = rsqrtf(var + 1e-5f);
  }
  __syncthreads();
  for (int i = tid; i < 16 * 128; i += 256) {
    int m = i >> 7, d = i & 127, row = it * 16 + m;
    out[((size_t)b * NNODE + row) * 128 + d] = (sy[m][d] - smu[m]) * srs[m] * lng[d] + lnb[d];
  }
}

// ---------------- context attention ----------------
__global__ __launch_bounds__(64) void ctx_score(const float* nf, const float* W1, const float* b1,
                                                const float* W2, const float* b2, float* scores) {
  int blk = blockIdx.x;             // 16*127
  int b = blk / NNBR, j = blk % NNBR, node = j + 1;
  int n = threadIdx.x;
  __shared__ float red[64];
  const float* x = nf + ((size_t)b * NNODE + node) * 128;
  float t = b1[n];
  for (int d = 0; d < 128; ++d) t += x[d] * W1[(size_t)n * 128 + d];
  t = fmaxf(t, 0.f);
  red[n] = t * W2[n]; __syncthreads();
  for (int o = 32; o > 0; o >>= 1) { if (n < o) red[n] += red[n + o]; __syncthreads(); }
  if (n == 0) scores[b * NNBR + j] = red[0] + b2[0];
}

__global__ __launch_bounds__(128) void ctx_combine(const float* nf, const float* scores, const int* mask,
                                                   float* context, _Float16* ctxfr) {
  int b = blockIdx.x, j = threadIdx.x;
  __shared__ float red[128];
  __shared__ float sattn[128];
  int keep = 0; float s = -1e30f;
  if (j < NNBR) { keep = mask[b * NNBR + j] != 0; if (keep) s = scores[b * NNBR + j]; }
  red[j] = s; __syncthreads();
  for (int o = 64; o > 0; o >>= 1) { if (j < o) red[j] = fmaxf(red[j], red[j + o]); __syncthreads(); }
  float mx = red[0]; __syncthreads();
  float e = (j < NNBR && keep) ? __expf(s - mx) : 0.f;
  red[j] = e; __syncthreads();
  for (int o = 64; o > 0; o >>= 1) { if (j < o) red[j] += red[j + o]; __syncthreads(); }
  float den = red[0]; if (den == 0.f) den = 1.f;
  sattn[j] = e / den; __syncthreads();
  int d = j;
  float c = nf[((size_t)b * NNODE + 0) * 128 + d];
  for (int n = 0; n < NNBR; ++n) c += sattn[n] * nf[((size_t)b * NNODE + 1 + n) * 128 + d];
  context[b * 128 + d] = c;
  ctxfr[(d >> 5) * 512 + afrag_idx(b, d & 31)] = (_Float16)c;   // A-frag rows = batches
}

// ---------------- heads: 6 mode MLPs + prob head (WMMA) ----------------
struct ModesP {
  const _Float16* pk1[6]; const float* b1[6];
  const _Float16* pk2[6]; const float* b2[6];
  const _Float16* pk3[6]; const float* b3[6];
};

__global__ __launch_bounds__(256) void heads(const _Float16* ctxfr, ModesP MP,
                                             const _Float16* pkp1, const float* pb1,
                                             const float* p2W, const float* p2b, float* out) {
  int blk = blockIdx.x;             // 0..5 modes, 6 = prob
  int tid = threadIdx.x, lane = tid & 31, wv = tid >> 5;
  __shared__ float sz[16][128];
  __shared__ alignas(64) _Float16 zfr[4 * 512];
  if (blk < 6) {
    // l1
    VF8 a1; a1.v = vz();
    for (int kk = 0; kk < 4; ++kk) {
      v16h a = ((const v16h*)(ctxfr + (size_t)kk * 512))[lane];
      v16h b = ((const v16h*)(MP.pk1[blk] + (size_t)(kk * 8 + wv) * 512))[lane];
      a1.v = wmma_f16(a, b, a1.v);
    }
    for (int i2 = 0; i2 < 8; ++i2) {
      int m = i2 + ((lane >> 4) << 3), n = lane & 15, c = wv * 16 + n;
      sz[m][c] = fmaxf(a1.f[i2] + MP.b1[blk][c], 0.f);
    }
    __syncthreads();
    for (int i = tid; i < 4 * 512; i += 256) { int kt = i >> 9, e = i & 511, m, k; inv_frag(e, m, k); zfr[i] = (_Float16)sz[m][kt * 32 + k]; }
    __syncthreads();
    // l2
    VF8 a2; a2.v = vz();
    for (int kk = 0; kk < 4; ++kk) {
      v16h a = ((const v16h*)(zfr + kk * 512))[lane];
      v16h b = ((const v16h*)(MP.pk2[blk] + (size_t)(kk * 8 + wv) * 512))[lane];
      a2.v = wmma_f16(a, b, a2.v);
    }
    for (int i2 = 0; i2 < 8; ++i2) {
      int m = i2 + ((lane >> 4) << 3), n = lane & 15, c = wv * 16 + n;
      sz[m][c] = fmaxf(a2.f[i2] + MP.b2[blk][c], 0.f);
    }
    __syncthreads();
    for (int i = tid; i < 4 * 512; i += 256) { int kt = i >> 9, e = i & 511, m, k; inv_frag(e, m, k); zfr[i] = (_Float16)sz[m][kt * 32 + k]; }
    __syncthreads();
    // l3 : N=60 padded to 64 (4 n-tiles, waves 0..3)
    if (wv < 4) {
      VF8 a3; a3.v = vz();
      for (int kk = 0; kk < 4; ++kk) {
        v16h a = ((const v16h*)(zfr + kk * 512))[lane];
        v16h b = ((const v16h*)(MP.pk3[blk] + (size_t)(kk * 4 + wv) * 512))[lane];
        a3.v = wmma_f16(a, b, a3.v);
      }
      for (int i2 = 0; i2 < 8; ++i2) {
        int m = i2 + ((lane >> 4) << 3), n = lane & 15, c = wv * 16 + n;
        if (c < TFUT * 2) {
          int tf = c >> 1, xy = c & 1;
          out[(((size_t)m * NMODE + blk) * TFUT + tf) * 2 + xy] = a3.f[i2] + MP.b3[blk][c];
        }
      }
    }
  } else {
    // prob1 (N=64, 4 tiles)
    if (wv < 4) {
      VF8 a1; a1.v = vz();
      for (int kk = 0; kk < 4; ++kk) {
        v16h a = ((const v16h*)(ctxfr + (size_t)kk * 512))[lane];
        v16h b = ((const v16h*)(pkp1 + (size_t)(kk * 4 + wv) * 512))[lane];
        a1.v = wmma_f16(a, b, a1.v);
      }
      for (int i2 = 0; i2 < 8; ++i2) {
        int m = i2 + ((lane >> 4) << 3), n = lane & 15, c = wv * 16 + n;
        sz[m][c] = fmaxf(a1.f[i2] + pb1[c], 0.f);
      }
    }
    __syncthreads();
    if (tid < 16) {
      int b = tid;
      float lg[6]; float mx = -1e30f;
      for (int k2 = 0; k2 < 6; ++k2) {
        float s = p2b[k2];
        for (int n2 = 0; n2 < 64; ++n2) s += sz[b][n2] * p2W[k2 * 64 + n2];
        lg[k2] = s; mx = fmaxf(mx, s);
      }
      float den = 0.f;
      for (int k2 = 0; k2 < 6; ++k2) { lg[k2] = __expf(lg[k2] - mx); den += lg[k2]; }
      for (int k2 = 0; k2 < 6; ++k2)
        out[(size_t)BATCH * NMODE * TFUT * 2 + b * 6 + k2] = lg[k2] / den;
    }
  }
}

// ---------------- host ----------------
extern "C" void kernel_launch(void* const* d_in, const int* in_sizes, int n_in,
                              void* d_out, int out_size, void* d_ws, size_t ws_size,
                              hipStream_t stream) {
  (void)in_sizes; (void)n_in; (void)out_size; (void)ws_size;
  auto F = [&](int i) { return (const float*)d_in[i]; };
  // Flatten order assumption: top-level insertion order {history, neighbors, neighbor_mask, params},
  // params flattened jax-style (dict keys sorted ASCII, lists in order, leaves {W<b} etc.)
  const float* history   = F(0);
  const float* neighbors = F(1);
  const int*   mask      = (const int*)d_in[2];
  // params: 3 ctx1.W, 4 ctx1.b, 5 ctx2.W, 6 ctx2.b
  // ego base 7:  bn1{b,g,rm,rv}=7..10  bn2=11..14  conv1{W,b}=15,16  conv2{W,b}=17,18  gru{Whh,Wih,bhh,bih}=19..22
  // gcn layer l base 23+10l: agg{W,b}, e1{W,b}, e2{W,b}, ln{b,g}, node{W,b}
  // modes i base 53+6i: l1{W,b}, l2{W,b}, l3{W,b}
  // nbr base 89 (same layout as ego);  105 prob1.W, 106 prob1.b, 107 prob2.W, 108 prob2.b, 109 type_emb
  const int EGO = 7, NBR = 89;

  char* ws = (char*)d_ws;
  size_t off = 0;
  auto alloc = [&](size_t bytes) { size_t o = off; off = (off + bytes + 1023) & ~(size_t)1023; return o; };
  auto aF16 = [&](size_t halves) { return (_Float16*)(ws + alloc(halves * 2)); };
  auto aF32 = [&](size_t elems)  { return (float*)(ws + alloc(elems * 4)); };

  _Float16* pk_wih_e = aF16(4 * 24 * 512);
  _Float16* pk_whh_e = aF16(4 * 24 * 512);
  _Float16* pk_wih_n = aF16(4 * 24 * 512);
  _Float16* pk_whh_n = aF16(4 * 24 * 512);
  _Float16* pk_c2_e  = aF16(6 * 8 * 512);
  _Float16* pk_c2_n  = aF16(6 * 8 * 512);
  _Float16* pk_proj[3]; _Float16* pk_agg[3];
  for (int l = 0; l < 3; ++l) { pk_proj[l] = aF16(4 * 24 * 512); pk_agg[l] = aF16(4 * 8 * 512); }
  _Float16* pk_l1[6]; _Float16* pk_l2[6]; _Float16* pk_l3[6];
  for (int i = 0; i < 6; ++i) { pk_l1[i] = aF16(4 * 8 * 512); pk_l2[i] = aF16(4 * 8 * 512); pk_l3[i] = aF16(4 * 4 * 512); }
  _Float16* pk_p1 = aF16(4 * 4 * 512);
  _Float16* xfrags = aF16((size_t)NGRP * THIST * 4 * 512);
  _Float16* HnFr   = aF16((size_t)BATCH * 4 * 8 * 512);
  _Float16* ctxfr  = aF16(4 * 512);
  float* hT     = aF32((size_t)NSEQ * HID);
  float* nfA    = aF32((size_t)BATCH * NNODE * HID);
  float* nfB    = aF32((size_t)BATCH * NNODE * HID);
  float* Pi     = aF32((size_t)BATCH * NNODE * HID);
  float* Pj     = aF32((size_t)BATCH * NNODE * HID);
  float* wsm    = aF32((size_t)BATCH * NNODE * NNODE);
  float* scores = aF32((size_t)BATCH * 128);
  float* context= aF32((size_t)BATCH * HID);

  auto packB = [&](const float* W, int ld, int koff, int K, int N, _Float16* dst, int NtTot, int nt0) {
    int Kt = (K + 31) / 32, Nt = (N + 15) / 16;
    pack_b<<<dim3(Kt * Nt), dim3(256), 0, stream>>>(W, ld, koff, K, N, dst, NtTot, nt0, Nt);
  };

  // ---- weight packing ----
  packB(F(EGO + 13), 128, 0, 128, 384, pk_wih_e, 24, 0);   // ego Wih
  packB(F(EGO + 12), 128, 0, 128, 384, pk_whh_e, 24, 0);   // ego Whh
  packB(F(NBR + 13), 128, 0, 128, 384, pk_wih_n, 24, 0);
  packB(F(NBR + 12), 128, 0, 128, 384, pk_whh_n, 24, 0);
  pack_conv2<<<dim3(48), dim3(256), 0, stream>>>(F(EGO + 10), pk_c2_e);
  pack_conv2<<<dim3(48), dim3(256), 0, stream>>>(F(NBR + 10), pk_c2_n);
  for (int l = 0; l < 3; ++l) {
    int g = 23 + 10 * l;
    packB(F(g + 8), 128, 0,   128, 128, pk_proj[l], 24, 0);    // node.W
    packB(F(g + 2), 259, 0,   128, 128, pk_proj[l], 24, 8);    // e1.W[:, :128]  -> Pi
    packB(F(g + 2), 259, 128, 128, 128, pk_proj[l], 24, 16);   // e1.W[:,128:256]-> Pj
    packB(F(g + 0), 128, 0,   128, 128, pk_agg[l], 8, 0);      // agg.W
  }
  for (int i = 0; i < 6; ++i) {
    int m = 53 + 6 * i;
    packB(F(m + 0), 128, 0, 128, 128, pk_l1[i], 8, 0);
    packB(F(m + 2), 128, 0, 128, 128, pk_l2[i], 8, 0);
    packB(F(m + 4), 128, 0, 128, 60,  pk_l3[i], 4, 0);
  }
  packB(F(105), 128, 0, 128, 64, pk_p1, 4, 0);

  // ---- encoders ----
  EncConvP ce{F(EGO + 8), F(EGO + 9), F(EGO + 1), F(EGO + 0), F(EGO + 2), F(EGO + 3),
              F(EGO + 11), F(EGO + 5), F(EGO + 4), F(EGO + 6), F(EGO + 7), pk_c2_e};
  EncConvP cn{F(NBR + 8), F(NBR + 9), F(NBR + 1), F(NBR + 0), F(NBR + 2), F(NBR + 3),
              F(NBR + 11), F(NBR + 5), F(NBR + 4), F(NBR + 6), F(NBR + 7), pk_c2_n};
  conv_encoder<<<dim3(NGRP), dim3(256), 0, stream>>>(history, neighbors, ce, cn, xfrags);

  GruP ge{pk_wih_e, pk_whh_e, F(EGO + 15), F(EGO + 14)};
  GruP gn{pk_wih_n, pk_whh_n, F(NBR + 15), F(NBR + 14)};
  gru_kernel<<<dim3(NGRP), dim3(256), 0, stream>>>(ge, gn, xfrags, hT);

  assemble_nodes<<<dim3((BATCH * NNODE * HID + 255) / 256), dim3(256), 0, stream>>>(hT, mask, F(109), nfA);

  // ---- GCN layers ----
  float* src = nfA; float* dst = nfB;
  for (int l = 0; l < 3; ++l) {
    int g = 23 + 10 * l;
    gcn_proj<<<dim3(128), dim3(256), 0, stream>>>(src, pk_proj[l], F(g + 9), F(g + 3), HnFr, Pi, Pj);
    gcn_edge<<<dim3(BATCH * NNODE), dim3(128), 0, stream>>>(Pi, Pj, F(g + 2), F(g + 4), F(g + 5),
                                                            mask, history, neighbors, wsm);
    gcn_msg<<<dim3(128), dim3(256), 0, stream>>>(wsm, HnFr, pk_agg[l], F(g + 1), src, F(g + 7), F(g + 6), dst);
    float* t = src; src = dst; dst = t;
  }
  const float* nf_final = src;  // after 3 swaps

  // ---- attention pooling ----
  ctx_score<<<dim3(BATCH * NNBR), dim3(64), 0, stream>>>(nf_final, F(3), F(4), F(5), F(6), scores);
  ctx_combine<<<dim3(BATCH), dim3(128), 0, stream>>>(nf_final, scores, mask, context, ctxfr);

  // ---- heads ----
  ModesP MP;
  for (int i = 0; i < 6; ++i) {
    int m = 53 + 6 * i;
    MP.pk1[i] = pk_l1[i]; MP.b1[i] = F(m + 1);
    MP.pk2[i] = pk_l2[i]; MP.b2[i] = F(m + 3);
    MP.pk3[i] = pk_l3[i]; MP.b3[i] = F(m + 5);
  }
  heads<<<dim3(7), dim3(256), 0, stream>>>(ctxfr, MP, pk_p1, F(106), F(107), F(108), (float*)d_out);
}
